// V10StyleEncoder_16587163697596
// MI455X (gfx1250) — matrix operand order repaired and checked by reading, not compiled, not run on hardware
//
#include <hip/hip_runtime.h>
#include <hip/hip_bf16.h>
#include <math.h>

// Problem constants (match reference)
#define BB     32
#define TT     2048
#define DINN   14
#define HH     512
#define CBN    64
#define NTOT   130
#define NBODY  128
#define MROWS  (BB*TT)     // 65536 frame rows
#define PROWS  (BB*NBODY)  // 4096 pooled rows

typedef __attribute__((ext_vector_type(8)))  __bf16 v8bf;
typedef __attribute__((ext_vector_type(16))) __bf16 v16bf;
typedef __attribute__((ext_vector_type(8)))  float  v8f;
typedef __attribute__((ext_vector_type(4)))  unsigned int v4u;
typedef __attribute__((ext_vector_type(8)))  int v8i;
typedef __attribute__((ext_vector_type(4)))  int v4i;

#if __has_builtin(__builtin_amdgcn_tensor_load_to_lds) && \
    __has_builtin(__builtin_amdgcn_s_wait_tensorcnt)
#define HAVE_TDM 1
#else
#define HAVE_TDM 0
#endif

__device__ __forceinline__ float wave_sum(float v) {
  #pragma unroll
  for (int off = 16; off > 0; off >>= 1) v += __shfl_xor(v, off, 32);
  return v;
}

__device__ __forceinline__ float gelu_tanh(float x) {
  const float k0 = 0.7978845608028654f, k1 = 0.044715f;
  return 0.5f * x * (1.0f + tanhf(k0 * (x + k1 * x * x * x)));
}

// ---------------------------------------------------------------------------
// Convert a (H,H) fp32 weight to bf16 transposed layout Wt[n*H + k] = W[k*H + n]
// so WMMA B-tiles become contiguous-K per lane (and TDM rows are contiguous).
// ---------------------------------------------------------------------------
__global__ void wt_bf16_kernel(const float* __restrict__ W, __bf16* __restrict__ Wt) {
  int idx = blockIdx.x * blockDim.x + threadIdx.x;  // n*H + k
  int n = idx >> 9, k = idx & (HH - 1);
  Wt[idx] = (__bf16)W[k * HH + n];
}

// ---------------------------------------------------------------------------
// x0 = LayerNorm(frames @ w_in + b_in).  One wave per row (DIN=14 is tiny).
// Writes fp32 X (residual path) and bf16 XB (WMMA A path).
// ---------------------------------------------------------------------------
__global__ void in_proj_ln_kernel(const float* __restrict__ frames,
                                  const float* __restrict__ w_in,
                                  const float* __restrict__ b_in,
                                  const float* __restrict__ g,
                                  const float* __restrict__ b,
                                  float* __restrict__ X, __bf16* __restrict__ XB) {
  int wave = threadIdx.x >> 5, lane = threadIdx.x & 31;
  int row = blockIdx.x * (blockDim.x >> 5) + wave;
  float f[DINN];
  #pragma unroll
  for (int d = 0; d < DINN; ++d) f[d] = frames[row * DINN + d];
  float v[16];
  float s = 0.f;
  #pragma unroll
  for (int j = 0; j < 16; ++j) {
    int col = j * 32 + lane;
    float a = b_in[col];
    #pragma unroll
    for (int d = 0; d < DINN; ++d) a = fmaf(f[d], w_in[d * HH + col], a);
    v[j] = a; s += a;
  }
  float mu = wave_sum(s) * (1.0f / HH);
  float vs = 0.f;
  #pragma unroll
  for (int j = 0; j < 16; ++j) { float d = v[j] - mu; vs += d * d; }
  float inv = rsqrtf(wave_sum(vs) * (1.0f / HH) + 1e-5f);
  #pragma unroll
  for (int j = 0; j < 16; ++j) {
    int col = j * 32 + lane;
    float y = (v[j] - mu) * inv * g[col] + b[col];
    X[row * HH + col] = y;
    XB[row * HH + col] = (__bf16)y;
  }
}

// ---------------------------------------------------------------------------
// WMMA GEMM:  Out = [Xres + gelu](Xb @ W + bias)
//   Xb : (M,512) bf16 row-major        (A matrices, from global + prefetch)
//   Wt : (512,512) bf16, Wt[n][k]      (B matrices; block's 64x512 slab is
//                                       DMA'd to LDS by the Tensor Data Mover)
// Each wave: one 16x16 tile, K-loop of 16 x v_wmma_f32_16x16x32_bf16.
// Block = 8 waves = 32 rows x 64 cols.  gridDim.x = (M/32) * (512/64).
// ---------------------------------------------------------------------------
__global__ void wmma_gemm_kernel(const __bf16* __restrict__ Xb,
                                 const __bf16* __restrict__ Wt,
                                 const float* __restrict__ bias,
                                 const float* __restrict__ Xres,
                                 float* __restrict__ Out) {
  __shared__ __bf16 sB[64 * HH];   // 64 cols x 512 K of weights = 64 KB (only LDS use)
  int wave = threadIdx.x >> 5, lane = threadIdx.x & 31;
  int bm = blockIdx.x >> 3;  // 8 column-blocks of 64
  int bn = blockIdx.x & 7;
  int row0 = bm * 32 + (wave >> 2) * 16;
  int colL = (wave & 3) * 16;          // column base inside the 64-wide slab
  int col0 = bn * 64 + colL;
  int la = lane & 15, lh = lane >> 4;

  const __bf16* gB = Wt + (size_t)(bn * 64) * HH;

#if HAVE_TDM
  // Flat-aperture rule (ISA 10.2): LDS generic address keeps the byte offset
  // in its low 32 bits, so this is sB's LDS offset for the D# descriptor.
  unsigned lds_off = (unsigned)(uintptr_t)(void*)sB;
  if (wave == 0) {
    // Tensor DMA descriptor (D#): 2D tile, 64 rows x 512 elems, 2-byte elems.
    unsigned long long ga = (unsigned long long)(uintptr_t)gB;
    v4u g0 = { 1u,                                  // count=1 (valid descriptor)
               lds_off,                             // lds_addr (bytes)
               (unsigned)(ga & 0xffffffffull),      // global_addr[31:0]
               (unsigned)((ga >> 32) & 0x1ffffffull) | (2u << 30) };  // addr[56:32] | type=2
    v8i g1 = { (int)(1u << 16),                     // data_size = 2 bytes
               (int)((HH & 0xffffu) << 16),         // tensor_dim0[15:0] @ bits 63:48
               (int)(((HH >> 16) & 0xffffu) | (64u << 16)),  // dim0 hi | tensor_dim1 lo
               (int)((HH & 0xffffu) << 16),         // dim1 hi (0) | tile_dim0 = 512
               (int)(64u),                          // tile_dim1 = 64 | tile_dim2 = 0
               (int)(HH),                           // tensor_dim0_stride[31:0] = 512
               0, 0 };
    v4i g2 = {0, 0, 0, 0}, g3 = {0, 0, 0, 0};
    v8i g4 = {0, 0, 0, 0, 0, 0, 0, 0};              // unused trailing group (clang-23 form)
    __builtin_amdgcn_tensor_load_to_lds(g0, g1, g2, g3, g4, 0);
    __builtin_amdgcn_s_wait_tensorcnt(0);
  }
  // The TDM writes sB behind the compiler's back; escape the pointer into an
  // opaque asm with a memory clobber so loads from sB are NOT folded to undef.
  asm volatile("" : : "v"(sB) : "memory");
  __syncthreads();
#else
  // Cooperative vectorized fallback: 256 threads fill the 64 KB slab.
  for (int i = threadIdx.x * 8; i < 64 * HH; i += 256 * 8)
    *(v8bf*)(sB + i) = *(const v8bf*)(gB + i);
  __syncthreads();
#endif

  // A-layout: lanes<16 hold K 0-15, lanes>=16 hold K 16-31 (matches B)
  const __bf16* arow = Xb + (size_t)(row0 + la) * HH + lh * 16;
  // B-layout: lanes<16 hold K 0-15 of column la, lanes>=16 hold K 16-31 (from LDS)
  const __bf16* brow = sB + (size_t)(colL + la) * HH + lh * 16;

  v8f acc = {};
  #pragma unroll
  for (int kt = 0; kt < HH; kt += 32) {
    union { v16bf v; v8bf h[2]; } A;
    A.h[0] = *(const v8bf*)(arow + kt);
    A.h[1] = *(const v8bf*)(arow + kt + 8);
    v16bf Bv = *(const v16bf*)(brow + kt);
    __builtin_prefetch(arow + kt + 64, 0, 1);   // global_prefetch_b8: A tile two K-steps ahead
    acc = __builtin_amdgcn_wmma_f32_16x16x32_bf16(
        /*neg_a=*/false, A.v, /*neg_b=*/false, Bv,
        /*c_mod=*/(short)0, acc, /*reuse_a=*/false, /*reuse_b=*/false);
  }

  // C/D layout: VGPR v -> row v (lanes 0-15) / row v+8 (lanes 16-31), col = lane&15
  int col = col0 + la;
  float bcol = bias[col];
  #pragma unroll
  for (int v = 0; v < 8; ++v) {
    int row = row0 + v + lh * 8;
    float val = acc[v] + bcol;
    if (Xres) val = Xres[(size_t)row * HH + col] + gelu_tanh(val);
    Out[(size_t)row * HH + col] = val;
  }
}

// ---------------------------------------------------------------------------
// Row LayerNorm: X = LN(Tm) (fp32 + bf16 copies); optionally attn scores.
// One wave per row.
// ---------------------------------------------------------------------------
__global__ void ln_rows_kernel(const float* __restrict__ Tm,
                               const float* __restrict__ g,
                               const float* __restrict__ b,
                               float* __restrict__ X, __bf16* __restrict__ XB,
                               const float* __restrict__ q,
                               float* __restrict__ scores) {
  int wave = threadIdx.x >> 5, lane = threadIdx.x & 31;
  int row = blockIdx.x * (blockDim.x >> 5) + wave;
  float v[16];
  float s = 0.f;
  #pragma unroll
  for (int j = 0; j < 16; ++j) {
    v[j] = Tm[(size_t)row * HH + j * 32 + lane];
    s += v[j];
  }
  float mu = wave_sum(s) * (1.0f / HH);
  float vs = 0.f;
  #pragma unroll
  for (int j = 0; j < 16; ++j) { float d = v[j] - mu; vs += d * d; }
  float inv = rsqrtf(wave_sum(vs) * (1.0f / HH) + 1e-5f);
  float sc = 0.f;
  #pragma unroll
  for (int j = 0; j < 16; ++j) {
    int col = j * 32 + lane;
    float y = (v[j] - mu) * inv * g[col] + b[col];
    X[(size_t)row * HH + col] = y;
    XB[(size_t)row * HH + col] = (__bf16)y;
    if (q) sc = fmaf(y, q[col], sc);
  }
  if (q) {
    sc = wave_sum(sc);
    if (lane == 0) scores[row] = sc * 0.04419417382415922f;  // * 512^-0.5
  }
}

// ---------------------------------------------------------------------------
// Alignment-masked softmax pooling.  One block per (b,n).  Deterministic:
// fixed thread->t assignment + fixed tree reductions + serial accumulate pass.
// ---------------------------------------------------------------------------
__global__ void pool_kernel(const float* __restrict__ X,
                            const float* __restrict__ scores,
                            const unsigned char* __restrict__ fmask,
                            const int* __restrict__ fpos,
                            __bf16* __restrict__ pooledB,
                            int* __restrict__ anyf) {
  int row = blockIdx.x;            // b*128 + n
  int b = row >> 7, n = row & 127;
  int target = n + 1;
  int tid = threadIdx.x;
  __shared__ float red[256];
  const unsigned char* mrow = fmask + (size_t)b * TT;
  const int* prow = fpos + (size_t)b * TT;
  const float* srow = scores + (size_t)b * TT;

  float lmax = -1e30f;
  for (int t = tid; t < TT; t += 256)
    if (mrow[t] && prow[t] == target) lmax = fmaxf(lmax, srow[t]);
  red[tid] = lmax; __syncthreads();
  for (int st = 128; st > 0; st >>= 1) {
    if (tid < st) red[tid] = fmaxf(red[tid], red[tid + st]);
    __syncthreads();
  }
  float gmax = red[0]; __syncthreads();
  int hasAny = (gmax > -1e29f);

  float lsum = 0.f;
  if (hasAny)
    for (int t = tid; t < TT; t += 256)
      if (mrow[t] && prow[t] == target) lsum += __expf(srow[t] - gmax);
  red[tid] = lsum; __syncthreads();
  for (int st = 128; st > 0; st >>= 1) {
    if (tid < st) red[tid] += red[tid + st];
    __syncthreads();
  }
  float gsum = red[0];

  float a0 = 0.f, a1 = 0.f;
  int c0 = tid, c1 = tid + 256;
  if (hasAny) {
    float inv = 1.0f / gsum;
    for (int t = 0; t < TT; ++t) {
      if (mrow[t] && prow[t] == target) {
        float w = __expf(srow[t] - gmax) * inv;
        const float* xr = X + ((size_t)b * TT + t) * HH;
        a0 = fmaf(w, xr[c0], a0);
        a1 = fmaf(w, xr[c1], a1);
      }
    }
  }
  pooledB[(size_t)row * HH + c0] = (__bf16)a0;
  pooledB[(size_t)row * HH + c1] = (__bf16)a1;
  if (tid == 0) anyf[row] = hasAny;
}

// ---------------------------------------------------------------------------
// VQ nearest code per row.  d2 computed directly as sum((e-c)^2), so the
// per-row minimum IS the commit-loss partial for that row.
// ---------------------------------------------------------------------------
__global__ void vq_kernel(const float* __restrict__ emb,
                          const float* __restrict__ cb,
                          int* __restrict__ codes,
                          float* __restrict__ rowsum) {
  int row = blockIdx.x;
  int tid = threadIdx.x;
  __shared__ float se[HH];
  __shared__ float pd[256];
  __shared__ float d2s[CBN];
  se[tid]       = emb[(size_t)row * HH + tid];
  se[tid + 256] = emb[(size_t)row * HH + tid + 256];
  __syncthreads();
  int c = tid & 63, seg = tid >> 6;               // 4 H-segments of 128
  const float* cr = cb + (size_t)c * HH + seg * 128;
  const float* er = se + seg * 128;
  float d = 0.f;
  #pragma unroll 4
  for (int h = 0; h < 128; ++h) { float df = er[h] - cr[h]; d = fmaf(df, df, d); }
  pd[tid] = d; __syncthreads();
  if (tid < 64) d2s[tid] = pd[tid] + pd[tid + 64] + pd[tid + 128] + pd[tid + 192];
  __syncthreads();
  if (tid == 0) {
    float best = d2s[0]; int bi = 0;
    for (int i = 1; i < CBN; ++i)
      if (d2s[i] < best) { best = d2s[i]; bi = i; }   // first-min, like argmin
    codes[row] = bi;
    rowsum[row] = best;
  }
}

// ---------------------------------------------------------------------------
// Deterministic commit reduction: 4096 row partials -> scalar.
// ---------------------------------------------------------------------------
__global__ void commit_kernel(const float* __restrict__ rowsum, float* __restrict__ out) {
  __shared__ float red[256];
  int tid = threadIdx.x;
  float s = 0.f;
  for (int i = 0; i < PROWS / 256; ++i) s += rowsum[tid * (PROWS / 256) + i];
  red[tid] = s; __syncthreads();
  for (int st = 128; st > 0; st >>= 1) {
    if (tid < st) red[tid] += red[tid + st];
    __syncthreads();
  }
  if (tid == 0) *out = 0.25f * red[0] / (float)((size_t)PROWS * HH);
}

// ---------------------------------------------------------------------------
// Assemble codes_full (B,130) and z_q_full (B,130,512).  z_q = quant (numeric
// value of straight-through) masked by any_frame; borders are PAD / zeros.
// ---------------------------------------------------------------------------
__global__ void assemble_kernel(const int* __restrict__ codes,
                                const int* __restrict__ anyf,
                                const float* __restrict__ cb,
                                float* __restrict__ out_codes,
                                float* __restrict__ out_z) {
  int bi = blockIdx.x;
  int b = bi / NTOT, i = bi % NTOT;
  int tid = threadIdx.x;
  int code = CBN, any = 0;
  if (i >= 1 && i <= NBODY) {
    int row = b * NBODY + (i - 1);
    any = anyf[row];
    code = any ? codes[row] : CBN;
  }
  float* z = out_z + ((size_t)b * NTOT + i) * HH;
  if (any) {
    z[tid]       = cb[(size_t)code * HH + tid];
    z[tid + 256] = cb[(size_t)code * HH + tid + 256];
  } else {
    z[tid] = 0.f; z[tid + 256] = 0.f;
  }
  if (tid == 0) out_codes[(size_t)b * NTOT + i] = (float)code;
}

// ---------------------------------------------------------------------------
extern "C" void kernel_launch(void* const* d_in, const int* in_sizes, int n_in,
                              void* d_out, int out_size, void* d_ws, size_t ws_size,
                              hipStream_t stream) {
  (void)in_sizes; (void)n_in; (void)out_size; (void)ws_size;
  const float* frames   = (const float*)d_in[0];
  const unsigned char* fmask = (const unsigned char*)d_in[1];  // jnp bool
  const int*   fpos     = (const int*)d_in[2];
  /* d_in[3] = n_total (130), fixed */
  const float* w_in  = (const float*)d_in[4];
  const float* b_in  = (const float*)d_in[5];
  const float* ln0_g = (const float*)d_in[6];
  const float* ln0_b = (const float*)d_in[7];
  const float* w1    = (const float*)d_in[8];
  const float* b1    = (const float*)d_in[9];
  const float* ln1_g = (const float*)d_in[10];
  const float* ln1_b = (const float*)d_in[11];
  const float* w2    = (const float*)d_in[12];
  const float* b2    = (const float*)d_in[13];
  const float* ln2_g = (const float*)d_in[14];
  const float* ln2_b = (const float*)d_in[15];
  const float* attnq = (const float*)d_in[16];
  const float* w_out = (const float*)d_in[17];
  const float* b_out = (const float*)d_in[18];
  const float* cb    = (const float*)d_in[19];

  // Workspace carve-up
  char* w = (char*)d_ws;
  float*  X   = (float*)w;   w += (size_t)MROWS * HH * 4;   // 128 MB  fp32 activations
  float*  Tb  = (float*)w;   w += (size_t)MROWS * HH * 4;   // 128 MB  pre-LN temp
  __bf16* XB  = (__bf16*)w;  w += (size_t)MROWS * HH * 2;   //  64 MB  bf16 activations
  __bf16* W1T = (__bf16*)w;  w += (size_t)HH * HH * 2;
  __bf16* W2T = (__bf16*)w;  w += (size_t)HH * HH * 2;
  __bf16* WOT = (__bf16*)w;  w += (size_t)HH * HH * 2;
  float*  SC  = (float*)w;   w += (size_t)MROWS * 4;        // attention scores (B,T)
  __bf16* PB  = (__bf16*)w;  w += (size_t)PROWS * HH * 2;   // pooled bf16
  float*  EMB = (float*)w;   w += (size_t)PROWS * HH * 4;   // emb fp32
  int*    ANY = (int*)w;     w += (size_t)PROWS * 4;
  int*    COD = (int*)w;     w += (size_t)PROWS * 4;
  float*  RS  = (float*)w;   w += (size_t)PROWS * 4;

  float* out_codes  = (float*)d_out;                          // (B,130)
  float* out_z      = out_codes + (size_t)BB * NTOT;          // (B,130,512)
  float* out_commit = out_z + (size_t)BB * NTOT * HH;         // scalar

  dim3 blk(256);

  // 1) bf16-transpose the three HxH weights
  wt_bf16_kernel<<<dim3(HH * HH / 256), blk, 0, stream>>>(w1, W1T);
  wt_bf16_kernel<<<dim3(HH * HH / 256), blk, 0, stream>>>(w2, W2T);
  wt_bf16_kernel<<<dim3(HH * HH / 256), blk, 0, stream>>>(w_out, WOT);

  // 2) input projection + LN0
  in_proj_ln_kernel<<<dim3(MROWS / 8), blk, 0, stream>>>(
      frames, w_in, b_in, ln0_g, ln0_b, X, XB);

  // 3) block 1: Tb = X + gelu(XB @ w1 + b1)   [WMMA + TDM]
  wmma_gemm_kernel<<<dim3((MROWS / 32) * 8), blk, 0, stream>>>(XB, W1T, b1, X, Tb);
  // 4) LN1
  ln_rows_kernel<<<dim3(MROWS / 8), blk, 0, stream>>>(Tb, ln1_g, ln1_b, X, XB,
                                                      nullptr, nullptr);
  // 5) block 2
  wmma_gemm_kernel<<<dim3((MROWS / 32) * 8), blk, 0, stream>>>(XB, W2T, b2, X, Tb);
  // 6) LN2 + attention scores
  ln_rows_kernel<<<dim3(MROWS / 8), blk, 0, stream>>>(Tb, ln2_g, ln2_b, X, XB,
                                                      attnq, SC);
  // 7) masked softmax pooling -> pooled bf16, any_frame
  pool_kernel<<<dim3(PROWS), blk, 0, stream>>>(X, SC, fmask, fpos, PB, ANY);

  // 8) out projection: EMB = PB @ w_out + b_out   [WMMA + TDM]
  wmma_gemm_kernel<<<dim3((PROWS / 32) * 8), blk, 0, stream>>>(PB, WOT, b_out,
                                                               nullptr, EMB);
  // 9) VQ argmin + commit partials
  vq_kernel<<<dim3(PROWS), blk, 0, stream>>>(EMB, cb, COD, RS);
  // 10) commit scalar
  commit_kernel<<<dim3(1), blk, 0, stream>>>(RS, out_commit);
  // 11) assemble outputs
  assemble_kernel<<<dim3(BB * NTOT), blk, 0, stream>>>(COD, ANY, cb,
                                                       out_codes, out_z);
}
